// RelativeMultiHeadedAttention_57174604644538
// MI455X (gfx1250) — matrix operand order repaired and checked by reading, not compiled, not run on hardware
//
#include <hip/hip_runtime.h>
#include <math.h>

// ---------------------------------------------------------------------------
// RelativeMultiHeadedAttention for MI455X (gfx1250, wave32, WMMA bf16 + TDM)
// B=2, S=2048, D=1024, H=16, DK=64
// Pipeline:
//   1) convert f32 inputs + weights -> bf16 (ws)
//   2) proj GEMM (bf16 WMMA, TDM-staged B): q,k,r -> [b,h,s,dk]; v -> [b,h,dk,s]
//   3) kpr = k + r (elementwise, packed bf16)
//   4) scores GEMM (bf16 WMMA, TDM-staged B): p = mask(q.kpr^T * 0.125)
//   5) softmax rows in place on p
//   6) out GEMM (bf16 WMMA, TDM-staged B, double-buffered): out = p . v
// ---------------------------------------------------------------------------

#define B_  2
#define S_  2048
#define D_  1024
#define H_  16
#define DK_ 64

typedef __bf16 bf16_t;
typedef __attribute__((ext_vector_type(16))) __bf16 v16bf;
typedef __attribute__((ext_vector_type(8)))  __bf16 v8bf;
typedef __attribute__((ext_vector_type(4)))  __bf16 v4bf;
typedef __attribute__((ext_vector_type(8)))  float  v8f;
typedef __attribute__((ext_vector_type(4)))  unsigned int u32x4;
typedef __attribute__((ext_vector_type(8)))  int i32x8;
typedef __attribute__((ext_vector_type(4)))  int i32x4;

#if defined(__has_builtin)
#if __has_builtin(__builtin_amdgcn_tensor_load_to_lds)
#define HAVE_TDM 1
#endif
#endif

#if defined(HAVE_TDM) && __has_include(<hip/amd_detail/amd_gfx1250_TDM.h>)
#define TDM_6ARG 1
#endif

__device__ __forceinline__ v8f zero_v8f() {
    v8f z;
#pragma unroll
    for (int i = 0; i < 8; ++i) z[i] = 0.0f;
    return z;
}

__device__ __forceinline__ v8f wmma_bf16(v16bf a, v16bf b, v8f c) {
    // D = A(16x32 bf16) * B(32x16 bf16) + C(16x16 f32)
    return __builtin_amdgcn_wmma_f32_16x16x32_bf16(
        /*neg_a=*/false, a, /*neg_b=*/false, b,
        /*c_mod=*/(short)0, c, /*reuse_a=*/false, /*reuse_b=*/false);
}

// A fragment: 16x32, lanes 0-15 hold K={k0..k0+7, k0+16..k0+23},
// lanes 16-31 hold K={k0+8..k0+15, k0+24..k0+31}; row = lane & 15.
__device__ __forceinline__ v16bf load_frag_a(const bf16_t* __restrict__ base,
                                             int ld, int k0, int lane) {
    const bf16_t* p = base + (size_t)(lane & 15) * ld + k0 + ((lane >> 4) << 3);
    v8bf lo = *(const v8bf*)p;
    v8bf hi = *(const v8bf*)(p + 16);
    return __builtin_shufflevector(lo, hi, 0, 1, 2, 3, 4, 5, 6, 7,
                                           8, 9, 10, 11, 12, 13, 14, 15);
}

// B fragment: 32x16, column n = memory row (lane & 15), same K order as A fragment.
__device__ __forceinline__ v16bf load_frag_b(const bf16_t* base, int ld, int k0, int lane) {
    const bf16_t* p = base + (size_t)(lane & 15) * ld + k0 + ((lane >> 4) << 3);
    v8bf lo = *(const v8bf*)p;
    v8bf hi = *(const v8bf*)(p + 16);
    return __builtin_shufflevector(lo, hi, 0, 1, 2, 3, 4, 5, 6, 7,
                                           8, 9, 10, 11, 12, 13, 14, 15);
}

// A fragment from f32 source (probability matrix), converted to bf16 on load.
__device__ __forceinline__ v16bf load_frag_a_f32(const float* __restrict__ base,
                                                 int ld, int k0, int lane) {
    const float* p = base + (size_t)(lane & 15) * ld + k0 + ((lane >> 4) << 3);
    float4 a0 = *(const float4*)(p + 0);
    float4 a1 = *(const float4*)(p + 4);
    float4 b0 = *(const float4*)(p + 16);
    float4 b1 = *(const float4*)(p + 20);
    v16bf r;
    r[0] = (bf16_t)a0.x; r[1] = (bf16_t)a0.y; r[2] = (bf16_t)a0.z; r[3] = (bf16_t)a0.w;
    r[4] = (bf16_t)a1.x; r[5] = (bf16_t)a1.y; r[6] = (bf16_t)a1.z; r[7] = (bf16_t)a1.w;
    r[8]  = (bf16_t)b0.x; r[9]  = (bf16_t)b0.y; r[10] = (bf16_t)b0.z; r[11] = (bf16_t)b0.w;
    r[12] = (bf16_t)b1.x; r[13] = (bf16_t)b1.y; r[14] = (bf16_t)b1.z; r[15] = (bf16_t)b1.w;
    return r;
}

// ---------------------------------------------------------------------------
// TDM: DMA a 64x64 bf16 tile (rows strided by ld_elems) into LDS.
// One issue per workgroup (wave 0); completion via TENSORcnt.
// ---------------------------------------------------------------------------
__device__ __forceinline__ unsigned lds_off(const void* p) {
    return (unsigned)(unsigned long long)p;   // low 32 bits of generic LDS addr
}

#ifdef HAVE_TDM
__device__ __forceinline__ void tdm_issue_2d(unsigned lds_byte, const void* gptr,
                                             unsigned long long tensor_d0,
                                             unsigned long long tensor_d1,
                                             unsigned long long stride0_elems) {
    const unsigned long long ga = (unsigned long long)gptr;
    u32x4 g0;
    g0[0] = 1u;                                            // count=1, user mode
    g0[1] = lds_byte;                                      // lds_addr (bytes)
    g0[2] = (unsigned)(ga & 0xffffffffu);                  // global_addr[31:0]
    g0[3] = (unsigned)((ga >> 32) & 0x1ffffffu) | (2u << 30); // addr[56:32], type=2
    i32x8 g1;
    g1[0] = (int)(1u << 16);                               // data_size=1 (2B), mask=0
    g1[1] = (int)((tensor_d0 & 0xffffu) << 16);            // tensor_dim0 lo16
    g1[2] = (int)(((tensor_d0 >> 16) & 0xffffu) |
                  ((tensor_d1 & 0xffffu) << 16));          // dim0 hi / dim1 lo
    g1[3] = (int)(((tensor_d1 >> 16) & 0xffffu) |
                  (64u << 16));                            // dim1 hi / tile_dim0=64
    g1[4] = (int)64;                                       // tile_dim1=64, tile_dim2=0
    g1[5] = (int)(stride0_elems & 0xffffffffu);            // dim0_stride lo32
    g1[6] = (int)((stride0_elems >> 32) & 0xffffu);        // dim0_stride hi16
    g1[7] = 0;
    i32x4 z4 = {0, 0, 0, 0};
#ifdef TDM_6ARG
    i32x8 z8 = {0, 0, 0, 0, 0, 0, 0, 0};
    __builtin_amdgcn_tensor_load_to_lds(g0, g1, z4, z4, z8, 0);
#else
    __builtin_amdgcn_tensor_load_to_lds(g0, g1, z4, z4, 0);
#endif
}
#endif

// Stage a 64x64 bf16 tile into LDS (rows packed at pitch 64).
__device__ __forceinline__ void stage_issue(bf16_t* lds, const bf16_t* g,
                                            int ld_elems,
                                            unsigned long long tensor_d0,
                                            unsigned long long tensor_d1) {
#ifdef HAVE_TDM
    if (threadIdx.x < 32) {
        tdm_issue_2d(lds_off(lds), (const void*)g, tensor_d0, tensor_d1,
                     (unsigned long long)ld_elems);
    }
#else
    for (int idx = threadIdx.x; idx < (64 * 64) / 8; idx += 256) {
        const int r = idx >> 3, c = (idx & 7) << 3;
        *(v8bf*)(lds + r * 64 + c) = *(const v8bf*)(g + (size_t)r * ld_elems + c);
    }
#endif
}

__device__ __forceinline__ void stage_wait() {
#ifdef HAVE_TDM
    if (threadIdx.x < 32) __builtin_amdgcn_s_wait_tensorcnt(0);
#endif
    __syncthreads();
}

// ---------------------------------------------------------------------------
// 1) f32 -> bf16 conversion
// ---------------------------------------------------------------------------
__global__ __launch_bounds__(256)
void cvt_f32_bf16_kernel(const float* __restrict__ src, bf16_t* __restrict__ dst, int n) {
    int i = (blockIdx.x * 256 + threadIdx.x) * 4;
    if (i < n) {
        float4 v = *(const float4*)(src + i);
        v4bf o;
        o[0] = (bf16_t)v.x; o[1] = (bf16_t)v.y; o[2] = (bf16_t)v.z; o[3] = (bf16_t)v.w;
        *(v4bf*)(dst + i) = o;
    }
}

// ---------------------------------------------------------------------------
// 2) Projection GEMM: Y[m][n] = sum_d X[m][d] * W[n][d] + bias[n]
//    M=4096, N=1024, K=1024. B tile (64 rows of W x 64 K) TDM-staged,
//    double-buffered in LDS.
// ---------------------------------------------------------------------------
__global__ __launch_bounds__(256)
void proj_gemm_kernel(const bf16_t* __restrict__ X,
                      const bf16_t* __restrict__ W,
                      const float*  __restrict__ bias,
                      bf16_t* __restrict__ Y,
                      int transpose_out) {
    __shared__ bf16_t ldsB[2][64 * 64];

    const int lane = threadIdx.x & 31;
    const int wave = threadIdx.x >> 5;
    const int m0 = blockIdx.x * 128 + (wave & 3) * 32;
    const int n0g = blockIdx.y * 64;               // block's N origin (global)
    const int nw = (wave >> 2) * 32;               // wave's N offset within block
    const int K = D_;

    v8f acc[2][2];
    acc[0][0] = zero_v8f(); acc[0][1] = zero_v8f();
    acc[1][0] = zero_v8f(); acc[1][1] = zero_v8f();

    const bf16_t* Xa0 = X + (size_t)m0 * K;
    const bf16_t* Xa1 = X + (size_t)(m0 + 16) * K;

    stage_issue(ldsB[0], W + (size_t)n0g * K, K, K, D_);
    const int nkt = K / 64;
    for (int kt = 0; kt < nkt; ++kt) {
        stage_wait();                                      // tile kt ready
        if (kt + 1 < nkt)
            stage_issue(ldsB[(kt + 1) & 1], W + (size_t)n0g * K + (kt + 1) * 64,
                        K, K, D_);
        const bf16_t* bt = ldsB[kt & 1];
#pragma unroll
        for (int ks = 0; ks < 64; ks += 32) {
            v16bf a0 = load_frag_a(Xa0, K, kt * 64 + ks, lane);
            v16bf a1 = load_frag_a(Xa1, K, kt * 64 + ks, lane);
            v16bf b0 = load_frag_b(bt + (size_t)nw * 64, 64, ks, lane);
            v16bf b1 = load_frag_b(bt + (size_t)(nw + 16) * 64, 64, ks, lane);
            acc[0][0] = wmma_bf16(a0, b0, acc[0][0]);
            acc[0][1] = wmma_bf16(a0, b1, acc[0][1]);
            acc[1][0] = wmma_bf16(a1, b0, acc[1][0]);
            acc[1][1] = wmma_bf16(a1, b1, acc[1][1]);
        }
    }

#pragma unroll
    for (int mi = 0; mi < 2; ++mi)
#pragma unroll
        for (int ni = 0; ni < 2; ++ni) {
            const int n = n0g + nw + ni * 16 + (lane & 15);
            const int mbase = m0 + mi * 16 + ((lane >> 4) << 3);
            const float bn = bias[n];
            const int h = n >> 6, dk = n & 63;
            v8f c = acc[mi][ni];
#pragma unroll
            for (int i = 0; i < 8; ++i) {
                const int m = mbase + i;
                const int b = m >> 11;          // S_ = 2048
                const int s = m & (S_ - 1);
                const float val = c[i] + bn;
                size_t idx = transpose_out
                    ? (((size_t)(b * H_ + h) * DK_ + dk) * S_ + s)
                    : (((size_t)(b * H_ + h) * S_ + s) * DK_ + dk);
                Y[idx] = (bf16_t)val;
            }
        }
}

// ---------------------------------------------------------------------------
// 3) kpr = k + r (packed bf16 elementwise)
// ---------------------------------------------------------------------------
__global__ __launch_bounds__(256)
void addkr_kernel(const bf16_t* __restrict__ k, const bf16_t* __restrict__ r,
                  bf16_t* __restrict__ kpr, int n) {
    int i = (blockIdx.x * 256 + threadIdx.x) * 8;
    if (i < n) {
        v8bf a = *(const v8bf*)(k + i);
        v8bf b = *(const v8bf*)(r + i);
        v8bf c;
#pragma unroll
        for (int j = 0; j < 8; ++j) c[j] = (bf16_t)((float)a[j] + (float)b[j]);
        *(v8bf*)(kpr + i) = c;
    }
}

// ---------------------------------------------------------------------------
// 4) Scores: P[bh][m][n] = mask(b,m,n) ? 0.125 * (q . kpr^T) : -1e9
//    Per (b,h): M=N=S=2048, K=DK=64. kpr tile (64 rows x 64) TDM-staged once.
// ---------------------------------------------------------------------------
__global__ __launch_bounds__(256)
void scores_kernel(const bf16_t* __restrict__ Q,
                   const bf16_t* __restrict__ KPR,
                   const int* __restrict__ mask,
                   float* __restrict__ P) {
    __shared__ bf16_t ldsB[64 * 64];

    const int bh = blockIdx.z;
    const int b  = bh >> 4;                 // H_ = 16
    const bf16_t* q  = Q   + (size_t)bh * S_ * DK_;
    const bf16_t* kp = KPR + (size_t)bh * S_ * DK_;
    const int* mrow = mask + (size_t)b * S_ * S_;
    float* prow = P + (size_t)bh * S_ * S_;

    const int lane = threadIdx.x & 31;
    const int wave = threadIdx.x >> 5;
    const int m0 = blockIdx.x * 128 + (wave & 3) * 32;
    const int n0g = blockIdx.y * 64;
    const int nw = (wave >> 2) * 32;

    stage_issue(ldsB, kp + (size_t)n0g * DK_, DK_, DK_, S_);

    v8f acc[2][2];
    acc[0][0] = zero_v8f(); acc[0][1] = zero_v8f();
    acc[1][0] = zero_v8f(); acc[1][1] = zero_v8f();

    const bf16_t* qa0 = q + (size_t)m0 * DK_;
    const bf16_t* qa1 = q + (size_t)(m0 + 16) * DK_;

    stage_wait();

#pragma unroll
    for (int k0 = 0; k0 < DK_; k0 += 32) {
        v16bf a0 = load_frag_a(qa0, DK_, k0, lane);
        v16bf a1 = load_frag_a(qa1, DK_, k0, lane);
        v16bf b0 = load_frag_b(ldsB + (size_t)nw * 64, 64, k0, lane);
        v16bf b1 = load_frag_b(ldsB + (size_t)(nw + 16) * 64, 64, k0, lane);
        acc[0][0] = wmma_bf16(a0, b0, acc[0][0]);
        acc[0][1] = wmma_bf16(a0, b1, acc[0][1]);
        acc[1][0] = wmma_bf16(a1, b0, acc[1][0]);
        acc[1][1] = wmma_bf16(a1, b1, acc[1][1]);
    }

    const float scale = 0.125f;             // 1/sqrt(64)
#pragma unroll
    for (int mi = 0; mi < 2; ++mi)
#pragma unroll
        for (int ni = 0; ni < 2; ++ni) {
            const int n = n0g + nw + ni * 16 + (lane & 15);
            const int mbase = m0 + mi * 16 + ((lane >> 4) << 3);
            v8f c = acc[mi][ni];
#pragma unroll
            for (int i = 0; i < 8; ++i) {
                const int m = mbase + i;
                float sc = c[i] * scale;
                if (mrow[(size_t)m * S_ + n] == 0) sc = -1.0e9f;
                prow[(size_t)m * S_ + n] = sc;
            }
        }
}

// ---------------------------------------------------------------------------
// 5) Row softmax in place. One block (8 waves) per row of 2048 floats.
// ---------------------------------------------------------------------------
__global__ __launch_bounds__(256)
void softmax_kernel(float* __restrict__ P) {
    __shared__ float red[256];
    float* p = P + (size_t)blockIdx.x * S_;
    const int t = threadIdx.x;

    float4 v0 = *(const float4*)(p + t * 8);
    float4 v1 = *(const float4*)(p + t * 8 + 4);
    float x[8] = {v0.x, v0.y, v0.z, v0.w, v1.x, v1.y, v1.z, v1.w};

    float lm = x[0];
#pragma unroll
    for (int i = 1; i < 8; ++i) lm = fmaxf(lm, x[i]);
    red[t] = lm;
    __syncthreads();
    for (int s = 128; s > 0; s >>= 1) {
        if (t < s) red[t] = fmaxf(red[t], red[t + s]);
        __syncthreads();
    }
    const float rowmax = red[0];
    __syncthreads();

    float ls = 0.0f;
    float e[8];
#pragma unroll
    for (int i = 0; i < 8; ++i) { e[i] = __expf(x[i] - rowmax); ls += e[i]; }
    red[t] = ls;
    __syncthreads();
    for (int s = 128; s > 0; s >>= 1) {
        if (t < s) red[t] += red[t + s];
        __syncthreads();
    }
    const float inv = 1.0f / red[0];

    float4 o0 = {e[0] * inv, e[1] * inv, e[2] * inv, e[3] * inv};
    float4 o1 = {e[4] * inv, e[5] * inv, e[6] * inv, e[7] * inv};
    *(float4*)(p + t * 8)     = o0;
    *(float4*)(p + t * 8 + 4) = o1;
}

// ---------------------------------------------------------------------------
// 6) out = p . v : per (b,h) M=2048, N=64, K=2048.
//    B tile (all 64 v^T rows x 64 K) TDM-staged, double-buffered.
// ---------------------------------------------------------------------------
__global__ __launch_bounds__(256)
void out_gemm_kernel(const float* __restrict__ P,
                     const bf16_t* __restrict__ VT,
                     float* __restrict__ O) {
    __shared__ bf16_t ldsB[2][64 * 64];

    const int bh = blockIdx.z;
    const float*  p  = P  + (size_t)bh * S_ * S_;
    const bf16_t* vt = VT + (size_t)bh * DK_ * S_;
    float* o = O + (size_t)bh * S_ * DK_;

    const int lane = threadIdx.x & 31;
    const int wave = threadIdx.x >> 5;
    const int m0 = blockIdx.x * 128 + (wave & 3) * 32;
    const int nw = (wave >> 2) * 32;      // N = 64 covered by one block

    v8f acc[2][2];
    acc[0][0] = zero_v8f(); acc[0][1] = zero_v8f();
    acc[1][0] = zero_v8f(); acc[1][1] = zero_v8f();

    const float* pa0 = p + (size_t)m0 * S_;
    const float* pa1 = p + (size_t)(m0 + 16) * S_;

    stage_issue(ldsB[0], vt, S_, S_, DK_);
    const int nkt = S_ / 64;
    for (int kt = 0; kt < nkt; ++kt) {
        stage_wait();                                   // tile kt ready
        if (kt + 1 < nkt)
            stage_issue(ldsB[(kt + 1) & 1], vt + (kt + 1) * 64, S_, S_, DK_);
        const bf16_t* bt = ldsB[kt & 1];
#pragma unroll
        for (int ks = 0; ks < 64; ks += 32) {
            const int k0 = kt * 64 + ks;
            // prefetch next K tile of the probability rows (streaming)
            __builtin_prefetch(pa0 + (size_t)(lane & 15) * S_ + k0 + 64, 0, 1);
            v16bf a0 = load_frag_a_f32(pa0, S_, k0, lane);
            v16bf a1 = load_frag_a_f32(pa1, S_, k0, lane);
            v16bf b0 = load_frag_b(bt + (size_t)nw * 64, 64, ks, lane);
            v16bf b1 = load_frag_b(bt + (size_t)(nw + 16) * 64, 64, ks, lane);
            acc[0][0] = wmma_bf16(a0, b0, acc[0][0]);
            acc[0][1] = wmma_bf16(a0, b1, acc[0][1]);
            acc[1][0] = wmma_bf16(a1, b0, acc[1][0]);
            acc[1][1] = wmma_bf16(a1, b1, acc[1][1]);
        }
    }

#pragma unroll
    for (int mi = 0; mi < 2; ++mi)
#pragma unroll
        for (int ni = 0; ni < 2; ++ni) {
            const int n = nw + ni * 16 + (lane & 15);
            const int mbase = m0 + mi * 16 + ((lane >> 4) << 3);
            v8f c = acc[mi][ni];
#pragma unroll
            for (int i = 0; i < 8; ++i) {
                o[(size_t)(mbase + i) * DK_ + n] = c[i];
            }
        }
}

// ---------------------------------------------------------------------------
// Host launcher
// ---------------------------------------------------------------------------
extern "C" void kernel_launch(void* const* d_in, const int* in_sizes, int n_in,
                              void* d_out, int out_size, void* d_ws, size_t ws_size,
                              hipStream_t stream) {
    (void)in_sizes; (void)n_in; (void)out_size; (void)ws_size;

    const float* query = (const float*)d_in[0];
    const float* key   = (const float*)d_in[1];
    const float* value = (const float*)d_in[2];
    const float* r     = (const float*)d_in[3];
    const int*   mask  = (const int*)d_in[4];
    const float* Wq = (const float*)d_in[5];
    const float* bq = (const float*)d_in[6];
    const float* Wk = (const float*)d_in[7];
    const float* bk = (const float*)d_in[8];
    const float* Wv = (const float*)d_in[9];
    const float* bv = (const float*)d_in[10];
    const float* Wr = (const float*)d_in[11];
    const float* br = (const float*)d_in[12];

    float* outO = (float*)d_out;                                  // [B,H,S,DK]
    float* outP = outO + (size_t)B_ * H_ * S_ * DK_;              // [B,H,S,S]

    const int nAct = B_ * S_ * D_;   // 4,194,304
    const int nW   = D_ * D_;        // 1,048,576

    char* ws = (char*)d_ws;
    size_t off = 0;
    auto suballoc = [&](size_t bytes) -> void* {
        void* ptr = (void*)(ws + off);
        off = (off + bytes + 255) & ~(size_t)255;
        return ptr;
    };

    bf16_t* xq  = (bf16_t*)suballoc((size_t)nAct * 2);
    bf16_t* xk  = (bf16_t*)suballoc((size_t)nAct * 2);
    bf16_t* xv  = (bf16_t*)suballoc((size_t)nAct * 2);
    bf16_t* xr  = (bf16_t*)suballoc((size_t)nAct * 2);
    bf16_t* wq  = (bf16_t*)suballoc((size_t)nW * 2);
    bf16_t* wk  = (bf16_t*)suballoc((size_t)nW * 2);
    bf16_t* wv  = (bf16_t*)suballoc((size_t)nW * 2);
    bf16_t* wr  = (bf16_t*)suballoc((size_t)nW * 2);
    bf16_t* qb  = (bf16_t*)suballoc((size_t)nAct * 2);   // [b,h,s,dk]
    bf16_t* kb  = (bf16_t*)suballoc((size_t)nAct * 2);   // [b,h,s,dk]
    bf16_t* rb  = (bf16_t*)suballoc((size_t)nAct * 2);   // [b,h,s,dk]
    bf16_t* vtb = (bf16_t*)suballoc((size_t)nAct * 2);   // [b,h,dk,s]
    bf16_t* kpr = (bf16_t*)suballoc((size_t)nAct * 2);   // [b,h,s,dk]

    // 1) conversions
    cvt_f32_bf16_kernel<<<nAct / 1024, 256, 0, stream>>>(query, xq, nAct);
    cvt_f32_bf16_kernel<<<nAct / 1024, 256, 0, stream>>>(key,   xk, nAct);
    cvt_f32_bf16_kernel<<<nAct / 1024, 256, 0, stream>>>(value, xv, nAct);
    cvt_f32_bf16_kernel<<<nAct / 1024, 256, 0, stream>>>(r,     xr, nAct);
    cvt_f32_bf16_kernel<<<nW / 1024, 256, 0, stream>>>(Wq, wq, nW);
    cvt_f32_bf16_kernel<<<nW / 1024, 256, 0, stream>>>(Wk, wk, nW);
    cvt_f32_bf16_kernel<<<nW / 1024, 256, 0, stream>>>(Wv, wv, nW);
    cvt_f32_bf16_kernel<<<nW / 1024, 256, 0, stream>>>(Wr, wr, nW);

    // 2) projections (M=4096, N=1024 -> grid 32 x 16)
    dim3 pg(32, 16, 1);
    proj_gemm_kernel<<<pg, 256, 0, stream>>>(xq, wq, bq, qb,  0);
    proj_gemm_kernel<<<pg, 256, 0, stream>>>(xk, wk, bk, kb,  0);
    proj_gemm_kernel<<<pg, 256, 0, stream>>>(xv, wv, bv, vtb, 1);
    proj_gemm_kernel<<<pg, 256, 0, stream>>>(xr, wr, br, rb,  0);

    // 3) kpr = k + r
    addkr_kernel<<<nAct / 2048, 256, 0, stream>>>(kb, rb, kpr, nAct);

    // 4) masked scaled scores into p_attn region
    scores_kernel<<<dim3(S_ / 128, S_ / 64, B_ * H_), 256, 0, stream>>>(qb, kpr, mask, outP);

    // 5) softmax rows in place
    softmax_kernel<<<B_ * H_ * S_, 256, 0, stream>>>(outP);

    // 6) out = p . v
    out_gemm_kernel<<<dim3(S_ / 128, 1, B_ * H_), 256, 0, stream>>>(outP, vtb, outO);
}
